// GCN_90838558311239
// MI455X (gfx1250) — matrix operand order repaired and checked by reading, not compiled, not run on hardware
//
#include <hip/hip_runtime.h>
#include <math.h>

// ---------- types ----------
typedef __bf16 bf16_t;
typedef __attribute__((ext_vector_type(16))) bf16_t v16bf;
typedef __attribute__((ext_vector_type(8)))  bf16_t v8bf;
typedef __attribute__((ext_vector_type(8)))  float  v8f;
typedef __attribute__((ext_vector_type(4)))  float  v4f;

#define KSPLIT 4

// ---------- kernel 1: dinv[i] = rsqrt(1 + sum_j adj[i,j]) ----------
__global__ __launch_bounds__(256) void gcn_rowsum_kernel(
    const float* __restrict__ adj, float* __restrict__ dinv, int N) {
  __shared__ float red[256];
  const int row = blockIdx.x;
  const float* p = adj + (size_t)row * N;
  float s = 0.0f;
  for (int j = threadIdx.x * 4; j < N; j += 256 * 4) {
    v4f v = *(const v4f*)(p + j);
    s += v.x + v.y + v.z + v.w;
  }
  red[threadIdx.x] = s;
  __syncthreads();
  for (int off = 128; off > 0; off >>= 1) {
    if (threadIdx.x < off) red[threadIdx.x] += red[threadIdx.x + off];
    __syncthreads();
  }
  if (threadIdx.x == 0) dinv[row] = rsqrtf(red[0] + 1.0f);
}

// ---------- kernel 2: xsT_hi/lo[f][j] = split_bf16(dinv[j] * x[j,f]) ----------
__global__ __launch_bounds__(256) void gcn_prep_kernel(
    const float* __restrict__ x, const float* __restrict__ dinv,
    unsigned short* __restrict__ xsT_hi, unsigned short* __restrict__ xsT_lo,
    int N, int F) {
  const int idx = blockIdx.x * 256 + threadIdx.x;   // idx = f*N + j
  if (idx >= N * F) return;
  const int f = idx / N;
  const int j = idx - f * N;
  const float v = dinv[j] * x[(size_t)j * F + f];
  const bf16_t hb = (bf16_t)v;                      // RNE
  const bf16_t lb = (bf16_t)(v - (float)hb);        // residual
  xsT_hi[idx] = __builtin_bit_cast(unsigned short, hb);
  xsT_lo[idx] = __builtin_bit_cast(unsigned short, lb);
}

// ---------- kernel 3: hpart[kc] = adj[:, kc-chunk] @ xs[kc-chunk, :] ----------
// Split-K x4. 4 waves/block, 16 rows/wave x 128 cols, K-step 32.
// B tile (hi+lo bf16, 16 KB) double-buffered in LDS via async-to-LDS DMA:
// the next tile's fill overlaps the current tile's 24 WMMAs.
__global__ __launch_bounds__(128) void gcn_gemm_kernel(
    const float* __restrict__ adj,
    const unsigned short* __restrict__ xsT_hi,
    const unsigned short* __restrict__ xsT_lo,
    float* __restrict__ hpart, int N, int F) {
  __shared__ __align__(16) unsigned short ldsB[2][2][128][32]; // [buf][hi/lo][f][k]
  const int tid    = threadIdx.x;
  const int lane   = tid & 31;
  const int wave   = tid >> 5;
  const int rBase  = blockIdx.x * 64 + wave * 16;
  const int hiHalf = lane >> 4;     // 0: lanes 0-15, 1: lanes 16-31
  const int lane15 = lane & 15;
  const int kChunk = N / KSPLIT;
  const int kStart = blockIdx.y * kChunk;
  const int kEnd   = kStart + kChunk;

  v8f acc[8] = {};                  // 8 C-tiles of 16x16 f32

  const float* arow = adj + (size_t)(rBase + lane15) * N;
  const unsigned int ldsBase = (unsigned int)(size_t)(&ldsB[0][0][0][0]);

  // async fill of one 16 KB B tile: thread t owns feature row f = t
  auto fillB = [&](int buf, int k0) {
    const unsigned short* srcHi = xsT_hi + (size_t)tid * N + k0; // 32 bf16 = 64 B
    const unsigned short* srcLo = xsT_lo + (size_t)tid * N + k0;
    const unsigned int base = ldsBase + (unsigned)(buf * 16384);
#pragma unroll
    for (int c = 0; c < 4; ++c) {
      unsigned long long gaH = (unsigned long long)(size_t)(srcHi + c * 8);
      unsigned int       laH = base + (unsigned)(tid * 64 + c * 16);
      asm volatile("global_load_async_to_lds_b128 %0, %1, off"
                   :: "v"(laH), "v"(gaH) : "memory");
      unsigned long long gaL = (unsigned long long)(size_t)(srcLo + c * 8);
      unsigned int       laL = base + (unsigned)(8192 + tid * 64 + c * 16);
      asm volatile("global_load_async_to_lds_b128 %0, %1, off"
                   :: "v"(laL), "v"(gaL) : "memory");
    }
  };

  int cur = 0;
  fillB(0, kStart);

  const int kb = hiHalf * 8;        // B: lanes 0-15 read K {0..7,16..23}, lanes 16-31 K {8..15,24..31}
  union BV { v8bf h2[2]; v16bf v; };

  for (int k0 = kStart; k0 < kEnd; k0 += 32) {
    // ---- issue A loads early: their HBM latency overlaps the wait+barrier ----
    // lane<16 holds K = k0+{0..7, 16..23}; lane>=16 holds K = k0+{8..15, 24..31}
    union { v4f f4[4]; float f[16]; } aw;
    const int ka = k0 + hiHalf * 8;
    aw.f4[0] = *(const v4f*)(arow + ka);
    aw.f4[1] = *(const v4f*)(arow + ka + 4);
    aw.f4[2] = *(const v4f*)(arow + ka + 16);
    aw.f4[3] = *(const v4f*)(arow + ka + 20);

    // ---- current B tile is ready once prior fill drains; single barrier ----
    asm volatile("s_wait_asynccnt 0x0" ::: "memory");
    __syncthreads();

    // ---- kick off next tile's fill; DMA overlaps this iteration's WMMAs ----
    if (k0 + 32 < kEnd) fillB(cur ^ 1, k0 + 32);

    // ---- split A tile (16x32 fp32 -> bf16 hi/lo) ----
    v16bf a_hi, a_lo;
#pragma unroll
    for (int e = 0; e < 16; ++e) {
      const float v = aw.f[e];
      const bf16_t hb = (bf16_t)v;
      a_hi[e] = hb;
      a_lo[e] = (bf16_t)(v - (float)hb);
    }

    // ---- compute: ping-pong B registers (prefetch tile t+1 during tile t) ----
    BV bh, bl, bh2, bl2;
    bh.h2[0] = *(const v8bf*)&ldsB[cur][0][lane15][kb];
    bh.h2[1] = *(const v8bf*)&ldsB[cur][0][lane15][kb + 16];
    bl.h2[0] = *(const v8bf*)&ldsB[cur][1][lane15][kb];
    bl.h2[1] = *(const v8bf*)&ldsB[cur][1][lane15][kb + 16];
#pragma unroll
    for (int t = 0; t < 8; ++t) {
      if (t < 7) {
        const int fn = (t + 1) * 16 + lane15;
        bh2.h2[0] = *(const v8bf*)&ldsB[cur][0][fn][kb];
        bh2.h2[1] = *(const v8bf*)&ldsB[cur][0][fn][kb + 16];
        bl2.h2[0] = *(const v8bf*)&ldsB[cur][1][fn][kb];
        bl2.h2[1] = *(const v8bf*)&ldsB[cur][1][fn][kb + 16];
      }
      acc[t] = __builtin_amdgcn_wmma_f32_16x16x32_bf16(
          false, a_hi, false, bh.v, (short)0, acc[t], false, false);
      acc[t] = __builtin_amdgcn_wmma_f32_16x16x32_bf16(
          false, a_lo, false, bh.v, (short)0, acc[t], false, false);
      acc[t] = __builtin_amdgcn_wmma_f32_16x16x32_bf16(
          false, a_hi, false, bl.v, (short)0, acc[t], false, false);
      bh = bh2;
      bl = bl2;
    }
    cur ^= 1;
  }

  // ---- write raw fp32 partials (dinv/identity folded in later) ----
  float* hp = hpart + (size_t)blockIdx.y * N * F;
#pragma unroll
  for (int t = 0; t < 8; ++t) {
#pragma unroll
    for (int v = 0; v < 8; ++v) {
      const int i = rBase + v + 8 * hiHalf;   // C/D: VGPR v -> M = v + 8*hiHalf
      const int f = t * 16 + lane15;
      hp[(size_t)i * F + f] = acc[t][v];
    }
  }
}

// ---------- kernel 4: reduce partials, apply dinv+identity, FC, normalize ----
__global__ __launch_bounds__(128) void gcn_fc_norm_kernel(
    const float* __restrict__ hpart, const float* __restrict__ x,
    const float* __restrict__ dinv, const float* __restrict__ W,
    const float* __restrict__ b, float* __restrict__ out, int N, int F) {
  __shared__ float hrow[128];
  __shared__ float red[128];
  const int i = blockIdx.x;
  const int t = threadIdx.x;
  float hv = 0.0f;
#pragma unroll
  for (int c = 0; c < KSPLIT; ++c) hv += hpart[((size_t)c * N + i) * F + t];
  const float di = dinv[i];
  hrow[t] = di * (hv + di * x[(size_t)i * F + t]);   // h = dinv.*(A@xs + xs)
  __syncthreads();
  float acc = b[t];
  const float* wr = W + (size_t)t * F;
#pragma unroll 8
  for (int k = 0; k < F; ++k) acc = fmaf(hrow[k], wr[k], acc);
  red[t] = acc * acc;
  __syncthreads();
  for (int off = 64; off > 0; off >>= 1) {
    if (t < off) red[t] += red[t + off];
    __syncthreads();
  }
  const float scale = 1.0f / fmaxf(sqrtf(red[0]), 1e-12f);
  out[(size_t)i * F + t] = acc * scale;
}

// ---------- launcher ----------
extern "C" void kernel_launch(void* const* d_in, const int* in_sizes, int n_in,
                              void* d_out, int out_size, void* d_ws, size_t ws_size,
                              hipStream_t stream) {
  const float* x   = (const float*)d_in[0];   // [N, F]
  const float* adj = (const float*)d_in[1];   // [N, N]
  const float* W   = (const float*)d_in[2];   // [F, F]
  const float* b   = (const float*)d_in[3];   // [F]
  const int F = in_sizes[3];                  // 128
  const int N = in_sizes[0] / F;              // 8192

  char* ws = (char*)d_ws;
  float*          dinv   = (float*)ws;                                    // N f32
  unsigned short* xsT_hi = (unsigned short*)(ws + 32768);                 // F*N bf16
  unsigned short* xsT_lo = (unsigned short*)(ws + 32768 + (size_t)N * F * 2);
  float*          hpart  = (float*)(ws + 32768 + (size_t)N * F * 4);      // KSPLIT*N*F f32

  gcn_rowsum_kernel<<<N, 256, 0, stream>>>(adj, dinv, N);
  gcn_prep_kernel<<<(N * F + 255) / 256, 256, 0, stream>>>(x, dinv, xsT_hi, xsT_lo, N, F);
  gcn_gemm_kernel<<<dim3(N / 64, KSPLIT), 128, 0, stream>>>(adj, xsT_hi, xsT_lo, hpart, N, F);
  gcn_fc_norm_kernel<<<N, 128, 0, stream>>>(hpart, x, dinv, W, b, (float*)d_out, N, F);
}
